// LSTMSubLayer_75385265979393
// MI455X (gfx1250) — matrix operand order repaired and checked by reading, compiled, not run on hardware
//
#include <hip/hip_runtime.h>
#include <hip/hip_bf16.h>

typedef __attribute__((ext_vector_type(16))) _Float16 v16h;
typedef __attribute__((ext_vector_type(8)))  _Float16 v8h;
typedef __attribute__((ext_vector_type(8)))  float    v8f;
typedef __attribute__((ext_vector_type(4)))  float    v4f;
typedef __attribute__((ext_vector_type(4)))  int      v4i;

typedef __attribute__((address_space(1))) v4i g_v4i;   // global int4
typedef __attribute__((address_space(3))) v4i l_v4i;   // LDS int4

#ifndef __has_builtin
#define __has_builtin(x) 0
#endif
#if __has_builtin(__builtin_amdgcn_global_load_async_to_lds_b128)
#define HAVE_ASYNC_LDS 1
#else
#define HAVE_ASYNC_LDS 0
#endif

__device__ __forceinline__ void wait_asynccnt0() {
#if __has_builtin(__builtin_amdgcn_s_wait_asynccnt)
    __builtin_amdgcn_s_wait_asynccnt(0);
#else
    asm volatile("s_wait_asynccnt 0x0" ::: "memory");
#endif
}

#define BATCH   64
#define SEQ     1024
#define INSZ    512
#define HD      512
#define KDIM    1024      // INSZ + HD
#define NSLICES 32        // hidden slices of 16 units
#define BGROUPS 4         // batch groups of 16 rows
#define NW      64        // gate columns per WG = 4 gates x 16 hidden

// ---------------------------------------------------------------------------
// Pack kernel: convert Wi/Wf/Wg/Wo (each [512 x 1024] f32) into f16 slice-major
// layout Wpack[slice s][j][k], j in 0..63 = (gate g = j/16, hidden h = s*16+j%16).
// Also packs biases and resets barrier counters (determinism across replays).
// ---------------------------------------------------------------------------
__global__ void lstm_pack(const float* __restrict__ Wi, const float* __restrict__ Wf,
                          const float* __restrict__ Wg, const float* __restrict__ Wo,
                          const float* __restrict__ bi, const float* __restrict__ bf,
                          const float* __restrict__ bg, const float* __restrict__ bo,
                          _Float16* __restrict__ Wpack, float* __restrict__ bpack,
                          unsigned* __restrict__ cnt) {
    int blk = blockIdx.x;            // 0..2047 -> (s, j)
    int s = blk >> 6, j = blk & 63;
    int g = j >> 4, hl = j & 15;
    int h = s * 16 + hl;
    const float* W = (g == 0) ? Wi : (g == 1) ? Wf : (g == 2) ? Wg : Wo;
    const float* B = (g == 0) ? bi : (g == 1) ? bf : (g == 2) ? bg : bo;
    const float* src = W + (size_t)h * KDIM;
    _Float16* dst = Wpack + (size_t)blk * KDIM;
    int k = threadIdx.x * 4;
    v4f w4 = *(const v4f*)(src + k);
    dst[k + 0] = (_Float16)w4.x;
    dst[k + 1] = (_Float16)w4.y;
    dst[k + 2] = (_Float16)w4.z;
    dst[k + 3] = (_Float16)w4.w;
    if (threadIdx.x == 0) bpack[blk] = B[h];
    if (blk == 0 && threadIdx.x < BGROUPS) cnt[threadIdx.x * 64] = 0u;
}

// ---------------------------------------------------------------------------
// Persistent LSTM kernel. Grid = 128 WGs (4 batch groups x 32 hidden slices),
// 128 threads = 4 wave32s. Weight slice (128 KB) stays in LDS for all 1024
// steps. Wave w computes gate w's 16x16 tile via v_wmma_f32_16x16x32_f16.
// ---------------------------------------------------------------------------
__device__ __forceinline__ void bg_barrier(unsigned* cnt, unsigned target) {
    __threadfence();
    __syncthreads();
    if (threadIdx.x == 0) {
        __hip_atomic_fetch_add(cnt, 1u, __ATOMIC_RELEASE, __HIP_MEMORY_SCOPE_AGENT);
        while (__hip_atomic_load(cnt, __ATOMIC_ACQUIRE, __HIP_MEMORY_SCOPE_AGENT) < target)
            __builtin_amdgcn_s_sleep(1);
    }
    __syncthreads();
}

__launch_bounds__(128, 1)
__global__ void lstm_persist(const float* __restrict__ x, const float* __restrict__ H0,
                             const float* __restrict__ C0,
                             const _Float16* __restrict__ Wpack,
                             const float* __restrict__ bpack,
                             _Float16* __restrict__ Hbuf, unsigned* __restrict__ cnt,
                             float* __restrict__ out) {
    __shared__ __attribute__((aligned(16))) _Float16 Wlds[NW][KDIM];    // 128 KB resident weights
    __shared__ __attribute__((aligned(16))) _Float16 combS[16][KDIM];   // 32 KB [x_t | H] staging
    __shared__ __attribute__((aligned(16))) float    gateS[4][16][16];  // 4 KB activated gates
    __shared__ __attribute__((aligned(16))) float    Cs[16][16];        // 1 KB cell state

    const int tid  = threadIdx.x;
    const int wave = tid >> 5;
    const int lane = tid & 31;
    const int n_lane = lane & 15;
    const int half   = lane >> 4;
    const int bg = blockIdx.x >> 5;       // batch group
    const int ns = blockIdx.x & 31;       // hidden slice
    const int b0 = bg * 16;
    const int h0 = ns * 16;
    unsigned* mycnt = cnt + bg * 64;

    // ---- Load resident weight slice: 64 rows x 1024 halfs = 128 KB ----
    {
        const uint4* gsrc = (const uint4*)(Wpack + (size_t)ns * NW * KDIM);
        uint4* ldst = (uint4*)&Wlds[0][0];
#if HAVE_ASYNC_LDS
        for (int i = tid; i < (NW * KDIM) / 8; i += 128)
            __builtin_amdgcn_global_load_async_to_lds_b128(
                (g_v4i*)(gsrc + i), (l_v4i*)(ldst + i), 0, 0);
        wait_asynccnt0();
#else
        for (int i = tid; i < (NW * KDIM) / 8; i += 128) ldst[i] = gsrc[i];
#endif
    }
    // ---- Init cell state tile and publish H0 slice (parity 0) ----
    for (int e = tid; e < 256; e += 128) {
        int m = e >> 4, n = e & 15;
        Cs[m][n] = C0[(size_t)(b0 + m) * HD + h0 + n];
        Hbuf[(size_t)(b0 + m) * HD + h0 + n] = (_Float16)H0[(size_t)(b0 + m) * HD + h0 + n];
    }
    unsigned epoch = 1;
    bg_barrier(mycnt, NSLICES * epoch);

    const float biasv = bpack[ns * 64 + wave * 16 + n_lane];
    int par = 0;

    for (int t = 0; t < SEQ; ++t) {
        // ---- Stage comb = [x_t (f32->f16) | H (f16, async L2->LDS)] ----
        {
            const int r = tid >> 3, c = tid & 7;   // 16 rows x 8 chunks of 64
            // H half: direct global->LDS async copy (no VGPR round trip)
            const _Float16* hsrc = Hbuf + ((size_t)par * BATCH + b0 + r) * HD + c * 64;
            _Float16* hdst = &combS[r][INSZ + c * 64];
#if HAVE_ASYNC_LDS
            #pragma unroll
            for (int i = 0; i < 8; ++i)
                __builtin_amdgcn_global_load_async_to_lds_b128(
                    (g_v4i*)(hsrc + i * 8), (l_v4i*)(hdst + i * 8), 0, 0);
#else
            #pragma unroll
            for (int i = 0; i < 8; ++i)
                ((uint4*)hdst)[i] = ((const uint4*)hsrc)[i];
#endif
            // x half: f32 -> f16 conversion (VALU work overlaps the async copy)
            const float* xsrc = x + ((size_t)(b0 + r) * SEQ + t) * INSZ + c * 64;
            _Float16* dstp = &combS[r][c * 64];
            #pragma unroll
            for (int i = 0; i < 16; ++i) {
                v4f v = *(const v4f*)(xsrc + i * 4);
                dstp[i * 4 + 0] = (_Float16)v.x;
                dstp[i * 4 + 1] = (_Float16)v.y;
                dstp[i * 4 + 2] = (_Float16)v.z;
                dstp[i * 4 + 3] = (_Float16)v.w;
            }
            if (t + 1 < SEQ) __builtin_prefetch(xsrc + INSZ, 0, 1);  // next x_t tile
#if HAVE_ASYNC_LDS
            wait_asynccnt0();
#endif
        }
        __syncthreads();

        // ---- GEMM: this wave's gate tile, M=16 N=16 K=1024 ----
        v8f acc;
        #pragma unroll
        for (int i = 0; i < 8; ++i) acc[i] = biasv;

        const int m = n_lane;                    // A-row for this lane
        const int jcol = wave * 16 + n_lane;     // B-column for this lane
        #pragma unroll
        for (int kk = 0; kk < KDIM / 32; ++kk) {
            const int k0 = kk * 32;
            // A fragment (16-bit A 16x32 layout: lane halves split K groups)
            v8h a0 = *(const v8h*)&combS[m][k0 + half * 8];
            v8h a1 = *(const v8h*)&combS[m][k0 + 16 + half * 8];
            v16h A = __builtin_shufflevector(a0, a1, 0, 1, 2, 3, 4, 5, 6, 7,
                                             8, 9, 10, 11, 12, 13, 14, 15);
            // B fragment (same K split as A: lane halves split K groups)
            v8h bb0 = *(const v8h*)&Wlds[jcol][k0 + half * 8];
            v8h bb1 = *(const v8h*)&Wlds[jcol][k0 + 16 + half * 8];
            v16h B = __builtin_shufflevector(bb0, bb1, 0, 1, 2, 3, 4, 5, 6, 7,
                                             8, 9, 10, 11, 12, 13, 14, 15);
            acc = __builtin_amdgcn_wmma_f32_16x16x32_f16(
                false, A, false, B, (short)0, acc, false, false);
        }

        // ---- Activation, share via LDS ----
        #pragma unroll
        for (int i = 0; i < 8; ++i) {
            float v = acc[i];
            float a = (wave == 2) ? tanhf(v) : (1.0f / (1.0f + __expf(-v)));
            gateS[wave][i + half * 8][n_lane] = a;
        }
        __syncthreads();

        // ---- Elementwise cell/hidden update + outputs ----
        const int nxt = par ^ 1;
        for (int e = tid; e < 256; e += 128) {
            int mm = e >> 4, nn = e & 15;
            float I = gateS[0][mm][nn];
            float F = gateS[1][mm][nn];
            float G = gateS[2][mm][nn];
            float O = gateS[3][mm][nn];
            float Cn = F * Cs[mm][nn] + I * G;
            float Hn = O * tanhf(Cn);
            Cs[mm][nn] = Cn;
            // write-once stream: non-temporal so it doesn't evict L2-resident x/H
            __builtin_nontemporal_store(Hn, &out[((size_t)(b0 + mm) * SEQ + t) * HD + h0 + nn]);
            Hbuf[((size_t)nxt * BATCH + b0 + mm) * HD + h0 + nn] = (_Float16)Hn;
            if (t == SEQ - 1) {
                out[(size_t)BATCH * SEQ * HD + (size_t)(b0 + mm) * HD + h0 + nn] = Hn;
                out[(size_t)BATCH * SEQ * HD + (size_t)BATCH * HD
                    + (size_t)(b0 + mm) * HD + h0 + nn] = Cn;
            }
        }

        ++epoch;
        bg_barrier(mycnt, NSLICES * epoch);
        par = nxt;
    }
}

// ---------------------------------------------------------------------------
// Host launch
// ---------------------------------------------------------------------------
extern "C" void kernel_launch(void* const* d_in, const int* in_sizes, int n_in,
                              void* d_out, int out_size, void* d_ws, size_t ws_size,
                              hipStream_t stream) {
    const float* x   = (const float*)d_in[0];
    const float* H0  = (const float*)d_in[1];
    const float* C0  = (const float*)d_in[2];
    const float* Wi  = (const float*)d_in[3];
    const float* bi  = (const float*)d_in[4];
    const float* Wf  = (const float*)d_in[5];
    const float* bf  = (const float*)d_in[6];
    const float* Wg  = (const float*)d_in[7];
    const float* bg  = (const float*)d_in[8];
    const float* Wo  = (const float*)d_in[9];
    const float* bo  = (const float*)d_in[10];
    float* out = (float*)d_out;

    char* ws = (char*)d_ws;
    _Float16* Wpack = (_Float16*)ws;                              // 4 MB
    float*    bpack = (float*)(ws + (4u << 20));                  // 8 KB
    unsigned* cnt   = (unsigned*)(ws + (4u << 20) + 8192);        // counters
    _Float16* Hbuf  = (_Float16*)(ws + (4u << 20) + 16384);       // 128 KB (x2 parity)

    lstm_pack<<<2048, 256, 0, stream>>>(Wi, Wf, Wg, Wo, bi, bf, bg, bo,
                                        Wpack, bpack, cnt);
    lstm_persist<<<128, 128, 0, stream>>>(x, H0, C0, Wpack, bpack, Hbuf, cnt, out);
}
